// Attention_77506979824184
// MI455X (gfx1250) — matrix operand order repaired and checked by reading, not compiled, not run on hardware
//
#include <hip/hip_runtime.h>
#include <hip/hip_bf16.h>

#define H 512
#define B 32
#define S 2048
#define K2H 1024           // 2*H
#define MROWS (B * S)      // 65536
#define APAD 1032          // 1024 + 8 bf16 padding (kills LDS bank conflicts)

typedef __bf16 v16bf __attribute__((ext_vector_type(16)));
typedef __bf16 v8bf  __attribute__((ext_vector_type(8)));
typedef float  v8f   __attribute__((ext_vector_type(8)));

union Frag16 { v16bf v; v8bf h[2]; };

__device__ __forceinline__ unsigned short f2bf(float f) {
  unsigned u = __float_as_uint(f);
  u += 0x7FFFu + ((u >> 16) & 1u);      // round-to-nearest-even
  return (unsigned short)(u >> 16);
}

// ---------------------------------------------------------------------------
// Kernel 1: dec_proj[b][h] = b_attn[h] + sum_k decoder_hide[b][k] * W_h[k][h]
// Tiny (16.8 MFLOP); plain VALU, coalesced over h, dec broadcast over lanes.
// ---------------------------------------------------------------------------
__global__ void dec_proj_kernel(const float* __restrict__ dh,
                                const float* __restrict__ W,   // W_attn (3H,H)
                                const float* __restrict__ bias,
                                float* __restrict__ dp) {
  int gid = blockIdx.x * blockDim.x + threadIdx.x;   // 0 .. B*H-1
  int b = gid >> 9;
  int h = gid & (H - 1);
  float sum = bias[h];
  const float* drow = dh + b * H;
#pragma unroll 8
  for (int k = 0; k < H; ++k)
    sum += drow[k] * W[k * H + h];
  dp[gid] = sum;
}

// ---------------------------------------------------------------------------
// Kernel 2: pack W_e (rows H..3H-1 of W_attn) into WMMA bf16 B-fragment layout.
// B (32x16, KxN) per 16x16x32 step: lanes 0-15 hold col n=lane, K=klo..klo+15;
// lanes 16-31 hold col n=lane-16, K=klo+16..klo+31. 16 bf16 (=8 uints)
// contiguous per lane -> coalesced dual b128 loads in the GEMM.
// packB[t][ks][lane][j]  t = n-tile (32), ks = k-step (32).
// ---------------------------------------------------------------------------
__global__ void packB_kernel(const float* __restrict__ W,       // W_attn
                             unsigned int* __restrict__ pb) {   // 256K uints
  int gid = blockIdx.x * blockDim.x + threadIdx.x;   // 0 .. 262143
  int j2   = gid & 7;
  int lane = (gid >> 3) & 31;
  int ks   = (gid >> 8) & 31;
  int t    = gid >> 13;
  int k0 = ks * 32 + ((lane >= 16) ? 16 : 0) + j2 * 2;
  int n  = t * 16 + (lane & 15);
  float f0 = W[(H + k0) * H + n];
  float f1 = W[(H + k0 + 1) * H + n];
  pb[gid] = (unsigned int)f2bf(f0) | ((unsigned int)f2bf(f1) << 16);
}

// ---------------------------------------------------------------------------
// Kernel 3: the big fused GEMM + tanh + v_w reduction + mask.
// One block (128 thr = 4 waves) per 16-row M-tile; block covers all N=512.
// ---------------------------------------------------------------------------
__global__ __launch_bounds__(128)
void attn_energy_kernel(const float* __restrict__ enc,          // (B,S,2H) f32
                        const unsigned short* __restrict__ pB,  // packed W_e
                        const float* __restrict__ dp,           // dec_proj (B,H)
                        const float* __restrict__ vw,           // v_w (H)
                        const int* __restrict__ maskp,          // (B,S)
                        float* __restrict__ out) {              // scores (B,S)
  __shared__ __align__(16) unsigned short sA[16 * APAD]; // 16 rows bf16, padded
  __shared__ float ssum[16];

  const int m    = blockIdx.x;          // m-tile index, 0..4095
  const int tid  = threadIdx.x;
  const int w    = tid >> 5;            // wave 0..3 -> n range [w*128, w*128+128)
  const int lane = tid & 31;
  const int b    = m >> 7;              // 128 m-tiles per batch row (S/16)

  if (tid < 16) ssum[tid] = 0.0f;

  // --- stage A: 16 rows x 1024 f32 -> bf16 in LDS (coalesced float4 loads) ---
  const long rowBase = (long)m * 16;
#pragma unroll
  for (int i = 0; i < 32; ++i) {
    int linear = i * 128 + tid;         // 0..4095 float4 chunks
    int row = linear >> 8;              // 256 float4 per row
    int c4  = linear & 255;
    const float4 f = *reinterpret_cast<const float4*>(
        enc + (rowBase + row) * (long)K2H + c4 * 4);
    unsigned int p0 = (unsigned int)f2bf(f.x) | ((unsigned int)f2bf(f.y) << 16);
    unsigned int p1 = (unsigned int)f2bf(f.z) | ((unsigned int)f2bf(f.w) << 16);
    uint2 pk; pk.x = p0; pk.y = p1;
    *reinterpret_cast<uint2*>(&sA[row * APAD + c4 * 4]) = pk;
  }
  __syncthreads();

  // --- WMMA K-loop: 32 steps x 8 n-subtiles, A fragment reused 8x ---
  v8f acc[8];
#pragma unroll
  for (int sub = 0; sub < 8; ++sub)
    acc[sub] = (v8f){0.f, 0.f, 0.f, 0.f, 0.f, 0.f, 0.f, 0.f};

  const int arow = lane & 15;
  const int ksel = (lane >> 4) << 4;    // lanes 16-31 hold K offsets +16..+31

  for (int ks = 0; ks < 32; ++ks) {
    const int klo = ks * 32 + ksel;
    const v8bf* pa = reinterpret_cast<const v8bf*>(&sA[arow * APAD + klo]);
    Frag16 a;
    a.h[0] = pa[0];                     // K = klo .. klo+7   (ds_load_b128)
    a.h[1] = pa[1];                     // K = klo+8 .. klo+15
#pragma unroll
    for (int sub = 0; sub < 8; ++sub) {
      const int tglob = w * 8 + sub;    // global n-tile 0..31
      const v8bf* pb = reinterpret_cast<const v8bf*>(
          pB + (((tglob * 32 + ks) * 32 + lane) << 4));
      Frag16 bf;
      bf.h[0] = pb[0];
      bf.h[1] = pb[1];
      acc[sub] = __builtin_amdgcn_wmma_f32_16x16x32_bf16(
          false, a.v, false, bf.v, (short)0, acc[sub], false, false);
    }
  }

  // --- epilogue: tanh(enc_proj + dec_proj) * v_w, reduce over h in-block ---
  // C/D layout: lane holds col n = lane&15; VGPR r -> row r (+8 for lanes>=16).
  float partial[8] = {0.f, 0.f, 0.f, 0.f, 0.f, 0.f, 0.f, 0.f};
  const int ncol = lane & 15;
#pragma unroll
  for (int sub = 0; sub < 8; ++sub) {
    const int h = w * 128 + sub * 16 + ncol;
    const float dpv = dp[b * H + h];
    const float vwv = vw[h];
#pragma unroll
    for (int r = 0; r < 8; ++r)
      partial[r] += tanhf(acc[sub][r] + dpv) * vwv;
  }
  const int rbase = (lane >> 4) << 3;
#pragma unroll
  for (int r = 0; r < 8; ++r)
    atomicAdd(&ssum[rbase + r], partial[r]);     // ds_add_f32
  __syncthreads();

  if (tid < 16) {
    const long gr = rowBase + tid;
    const float sc = ssum[tid];
    out[gr] = (maskp[gr] == 0) ? -100000.0f : sc;
  }
}

// ---------------------------------------------------------------------------
// Kernel 4: row softmax over S=2048, in place on d_out. One block per batch b.
// ---------------------------------------------------------------------------
__global__ void softmax_kernel(float* __restrict__ out) {
  __shared__ float red[256];
  const int b = blockIdx.x;
  const int tid = threadIdx.x;
  float v[8];
  float mx = -3.402823e38f;
#pragma unroll
  for (int i = 0; i < 8; ++i) {
    v[i] = out[b * S + i * 256 + tid];
    mx = fmaxf(mx, v[i]);
  }
  red[tid] = mx;
  __syncthreads();
  for (int s = 128; s > 0; s >>= 1) {
    if (tid < s) red[tid] = fmaxf(red[tid], red[tid + s]);
    __syncthreads();
  }
  mx = red[0];
  __syncthreads();
  float sum = 0.f;
#pragma unroll
  for (int i = 0; i < 8; ++i) {
    v[i] = __expf(v[i] - mx);
    sum += v[i];
  }
  red[tid] = sum;
  __syncthreads();
  for (int s = 128; s > 0; s >>= 1) {
    if (tid < s) red[tid] += red[tid + s];
    __syncthreads();
  }
  const float inv = 1.0f / red[0];
#pragma unroll
  for (int i = 0; i < 8; ++i)
    out[b * S + i * 256 + tid] = v[i] * inv;
}

// ---------------------------------------------------------------------------
extern "C" void kernel_launch(void* const* d_in, const int* in_sizes, int n_in,
                              void* d_out, int out_size, void* d_ws, size_t ws_size,
                              hipStream_t stream) {
  const float* decoder_hide = (const float*)d_in[0];   // (B,H)
  const float* encoder_out  = (const float*)d_in[1];   // (B,S,2H)
  const int*   mask         = (const int*)d_in[2];     // (B,S)
  const float* W_attn       = (const float*)d_in[3];   // (3H,H)
  const float* b_attn       = (const float*)d_in[4];   // (H)
  const float* v_w          = (const float*)d_in[5];   // (H)
  float* out = (float*)d_out;                          // (B,S)

  // workspace: [0,1MB) packed bf16 W_e ; [1MB, 1MB+64KB) dec_proj f32
  unsigned short* pB = (unsigned short*)d_ws;
  float* dp = (float*)((char*)d_ws + (1u << 20));

  // independent prep kernels (stream-ordered before the GEMM)
  dec_proj_kernel<<<(B * H) / 256, 256, 0, stream>>>(decoder_hide, W_attn,
                                                     b_attn, dp);
  packB_kernel<<<1024, 256, 0, stream>>>(W_attn, (unsigned int*)pB);

  // fused GEMM + tanh + v_w + mask: 4096 M-tiles of 16 rows
  attn_energy_kernel<<<MROWS / 16, 128, 0, stream>>>(encoder_out, pB, dp, v_w,
                                                     mask, out);

  // row softmax in place
  softmax_kernel<<<B, 256, 0, stream>>>(out);
}
